// MLAAttention_39230231282051
// MI455X (gfx1250) — compile-verified, matrix operand order checked by reading, not run on hardware
//
#include <hip/hip_runtime.h>
#include <stddef.h>

// ---------------- problem constants ----------------
static constexpr int Bc   = 4;
static constexpr int Tc   = 2048;
static constexpr int Hc   = 8;
static constexpr int KVHc = 2;
static constexpr int MR   = Bc * Tc;      // 8192 rows
static constexpr float LOG_ROPE_BASE = 13.122363377404328f;   // ln(500000)

typedef __attribute__((ext_vector_type(16))) __bf16 v16bf;
typedef __attribute__((ext_vector_type(8)))  __bf16 v8bf;
typedef __attribute__((ext_vector_type(8)))  float  v8f;

// ---------------- small helpers ----------------
__device__ __forceinline__ __bf16 f2bf(float f) {
  union { float f; unsigned u; } x; x.f = f;
  unsigned r = (x.u + 0x7FFFu + ((x.u >> 16) & 1u)) >> 16;
  unsigned short h = (unsigned short)r;
  __bf16 out;
  __builtin_memcpy(&out, &h, 2);
  return out;
}

__device__ __forceinline__ float red_add16(float v) {
  v += __shfl_xor(v, 1, 32);
  v += __shfl_xor(v, 2, 32);
  v += __shfl_xor(v, 4, 32);
  v += __shfl_xor(v, 8, 32);
  return v;
}
__device__ __forceinline__ float red_max16(float v) {
  v = fmaxf(v, __shfl_xor(v, 1, 32));
  v = fmaxf(v, __shfl_xor(v, 2, 32));
  v = fmaxf(v, __shfl_xor(v, 4, 32));
  v = fmaxf(v, __shfl_xor(v, 8, 32));
  return v;
}

__device__ __forceinline__ v8f wmma_bf16(v16bf a, v16bf b, v8f c) {
  return __builtin_amdgcn_wmma_f32_16x16x32_bf16(false, a, false, b, (short)0, c,
                                                 false, false);
}

// Unified 16x32 fragment loader. Source is row-major with K contiguous per row:
//  - A operand: row = M index
//  - B operand: row = N index over a transposed matrix (W^T, K^T, V^T)
// Per-lane data is two contiguous 16-byte chunks -> _b128 loads.
// slot layout (ISA 7.12.2): lane<16 K {0..7,16..23}; lane>=16 {8..15,24..31}
__device__ __forceinline__ v16bf load_frag(const __bf16* base, int ldr,
                                           int rbase, int kbase, int lane) {
  const __bf16* p = base + (size_t)(rbase + (lane & 15)) * ldr + kbase
                         + ((lane < 16) ? 0 : 8);
  v8bf lo = *(const v8bf*)(p);
  v8bf hi = *(const v8bf*)(p + 16);
  v16bf r;
  __builtin_memcpy(&r, &lo, 16);
  __builtin_memcpy((char*)&r + 16, &hi, 16);
  return r;
}

// Same, with row clamp (attention key tiles that overrun T)
__device__ __forceinline__ v16bf load_frag_clamp(const __bf16* base, int ldr,
                                                 int rbase, int kbase, int rmax,
                                                 int lane) {
  int rr = rbase + (lane & 15); if (rr > rmax) rr = rmax;
  const __bf16* p = base + (size_t)rr * ldr + kbase + ((lane < 16) ? 0 : 8);
  v8bf lo = *(const v8bf*)(p);
  v8bf hi = *(const v8bf*)(p + 16);
  v16bf r;
  __builtin_memcpy(&r, &lo, 16);
  __builtin_memcpy((char*)&r + 16, &hi, 16);
  return r;
}

// Register-resident partial RoPE on two 16-wide N-tiles (dims 0..15 / 16..31)
__device__ __forceinline__ void apply_rope(v8f& x1v, v8f& x2v, int rowbase, int lane) {
  float inv = __expf(-(float)(lane & 15) * (1.0f / 16.0f) * LOG_ROPE_BASE);
  int half = (lane < 16) ? 0 : 8;
#pragma unroll
  for (int j = 0; j < 8; ++j) {
    int t = (rowbase + j + half) & (Tc - 1);
    float ang = (float)t * inv;
    float cs = __cosf(ang), sn = __sinf(ang);
    float x1 = x1v[j], x2 = x2v[j];
    x1v[j] = x1 * cs - x2 * sn;
    x2v[j] = x2 * cs + x1 * sn;
  }
}

// RMSNorm over a 64-wide head row spread across 4 C fragments
__device__ __forceinline__ void head_rms4(v8f& a0, v8f& a1, v8f& a2, v8f& a3,
                                          const float* __restrict__ w, int lane) {
  int c = lane & 15;
  float w0 = w[c], w1 = w[16 + c], w2 = w[32 + c], w3 = w[48 + c];
#pragma unroll
  for (int j = 0; j < 8; ++j) {
    float ss = a0[j]*a0[j] + a1[j]*a1[j] + a2[j]*a2[j] + a3[j]*a3[j];
    ss = red_add16(ss);
    float r = rsqrtf(ss * (1.0f / 64.0f) + 1e-6f);
    a0[j] *= r * w0; a1[j] *= r * w1; a2[j] *= r * w2; a3[j] *= r * w3;
  }
}

// ---------------- staging kernels ----------------
__global__ void cast_f32_to_bf16(const float* __restrict__ src,
                                 __bf16* __restrict__ dst, int n) {
  int i = blockIdx.x * blockDim.x + threadIdx.x;
  int stride = gridDim.x * blockDim.x;
  for (; i < n; i += stride) dst[i] = f2bf(src[i]);
}

// src [K][N] f32 -> dst [N][K] bf16 (weights: make B-fragments K-contiguous)
__global__ void transpose_cast_kernel(const float* __restrict__ src,
                                      __bf16* __restrict__ dst, int K, int N) {
  int i = blockIdx.x * blockDim.x + threadIdx.x;
  int total = K * N;
  int stride = gridDim.x * blockDim.x;
  for (; i < total; i += stride) {
    int k = i / N, n = i % N;
    dst[(size_t)n * K + k] = f2bf(src[i]);
  }
}

// ---------------- compute kernels ----------------
// C = RMSNorm(Xbf[MR,512] @ W) * lnw -> bf16 [MR,N]; WT is [N][512]
template <int N>
__global__ void rms_down_kernel(const __bf16* __restrict__ X,
                                const __bf16* __restrict__ WT,
                                const float* __restrict__ lnw,
                                __bf16* __restrict__ C) {
  constexpr int NT = N / 16;
  int wid  = blockIdx.x * (blockDim.x >> 5) + (threadIdx.x >> 5);
  int lane = threadIdx.x & 31;
  if (wid * 16 >= MR) return;
  int mbase = wid * 16;

  v8f zero = {0.f,0.f,0.f,0.f,0.f,0.f,0.f,0.f};
  v8f acc[NT];
#pragma unroll
  for (int n = 0; n < NT; ++n) acc[n] = zero;

  for (int kc = 0; kc < 512; kc += 32) {
    v16bf a = load_frag(X, 512, mbase, kc, lane);
#pragma unroll
    for (int n = 0; n < NT; ++n)
      acc[n] = wmma_bf16(a, load_frag(WT, 512, n * 16, kc, lane), acc[n]);
  }

  int c = lane & 15, half = (lane < 16) ? 0 : 8;
  float wv[NT];
#pragma unroll
  for (int n = 0; n < NT; ++n) wv[n] = lnw[n * 16 + c];
#pragma unroll
  for (int j = 0; j < 8; ++j) {
    float ss = 0.f;
#pragma unroll
    for (int n = 0; n < NT; ++n) ss += acc[n][j] * acc[n][j];
    ss = red_add16(ss);
    float r = rsqrtf(ss * (1.0f / (float)N) + 1e-6f);
    size_t row = (size_t)(mbase + j + half) * N;
#pragma unroll
    for (int n = 0; n < NT; ++n)
      C[row + n * 16 + c] = f2bf(acc[n][j] * r * wv[n]);
  }
}

// q = head_rms(concat(c_q@W_UQ, rope(c_q@W_QR))) -> bf16 [B][H][T][64]
__global__ void q_proj_kernel(const __bf16* __restrict__ Cq,
                              const __bf16* __restrict__ WuqT,  // [256][256]
                              const __bf16* __restrict__ WqrT,  // [256][256]
                              const float* __restrict__ hw,
                              __bf16* __restrict__ Qout) {
  int wid  = blockIdx.x * (blockDim.x >> 5) + (threadIdx.x >> 5);
  int lane = threadIdx.x & 31;
  if (wid >= (MR / 16) * Hc) return;
  int h = wid % Hc, rt = wid / Hc;
  int mbase = rt * 16;

  v8f zero = {0.f,0.f,0.f,0.f,0.f,0.f,0.f,0.f};
  v8f a0 = zero, a1 = zero, a2 = zero, a3 = zero;
  for (int kc = 0; kc < 256; kc += 32) {
    v16bf a = load_frag(Cq, 256, mbase, kc, lane);
    a0 = wmma_bf16(a, load_frag(WuqT, 256, h * 32,      kc, lane), a0);
    a1 = wmma_bf16(a, load_frag(WuqT, 256, h * 32 + 16, kc, lane), a1);
    a2 = wmma_bf16(a, load_frag(WqrT, 256, h * 32,      kc, lane), a2);
    a3 = wmma_bf16(a, load_frag(WqrT, 256, h * 32 + 16, kc, lane), a3);
  }
  apply_rope(a2, a3, mbase, lane);
  head_rms4(a0, a1, a2, a3, hw, lane);

  int b = mbase / Tc, t0 = mbase % Tc;
  __bf16* dst = Qout + ((size_t)(b * Hc + h) * Tc + t0) * 64;
  int c = lane & 15, half = (lane < 16) ? 0 : 8;
#pragma unroll
  for (int j = 0; j < 8; ++j) {
    size_t r = (size_t)(j + half) * 64;
    dst[r +  0 + c] = f2bf(a0[j]);
    dst[r + 16 + c] = f2bf(a1[j]);
    dst[r + 32 + c] = f2bf(a2[j]);
    dst[r + 48 + c] = f2bf(a3[j]);
  }
}

// k = head_rms(concat(c_kv@W_UK, rope(x@W_KR))) -> bf16 [B][KVH][T][64]
__global__ void k_proj_kernel(const __bf16* __restrict__ Ckv,
                              const __bf16* __restrict__ Xbf,
                              const __bf16* __restrict__ WukT,  // [64][128]
                              const __bf16* __restrict__ WkrT,  // [64][512]
                              const float* __restrict__ hw,
                              __bf16* __restrict__ Kout) {
  int wid  = blockIdx.x * (blockDim.x >> 5) + (threadIdx.x >> 5);
  int lane = threadIdx.x & 31;
  if (wid >= (MR / 16) * KVHc) return;
  int kvh = wid % KVHc, rt = wid / KVHc;
  int mbase = rt * 16;

  v8f zero = {0.f,0.f,0.f,0.f,0.f,0.f,0.f,0.f};
  v8f a0 = zero, a1 = zero, a2 = zero, a3 = zero;
  for (int kc = 0; kc < 128; kc += 32) {
    v16bf a = load_frag(Ckv, 128, mbase, kc, lane);
    a0 = wmma_bf16(a, load_frag(WukT, 128, kvh * 32,      kc, lane), a0);
    a1 = wmma_bf16(a, load_frag(WukT, 128, kvh * 32 + 16, kc, lane), a1);
  }
  for (int kc = 0; kc < 512; kc += 32) {
    v16bf a = load_frag(Xbf, 512, mbase, kc, lane);
    a2 = wmma_bf16(a, load_frag(WkrT, 512, kvh * 32,      kc, lane), a2);
    a3 = wmma_bf16(a, load_frag(WkrT, 512, kvh * 32 + 16, kc, lane), a3);
  }
  apply_rope(a2, a3, mbase, lane);
  head_rms4(a0, a1, a2, a3, hw, lane);

  int b = mbase / Tc, t0 = mbase % Tc;
  __bf16* dst = Kout + ((size_t)(b * KVHc + kvh) * Tc + t0) * 64;
  int c = lane & 15, half = (lane < 16) ? 0 : 8;
#pragma unroll
  for (int j = 0; j < 8; ++j) {
    size_t r = (size_t)(j + half) * 64;
    dst[r +  0 + c] = f2bf(a0[j]);
    dst[r + 16 + c] = f2bf(a1[j]);
    dst[r + 32 + c] = f2bf(a2[j]);
    dst[r + 48 + c] = f2bf(a3[j]);
  }
}

// v = c_kv @ W_UV, stored TRANSPOSED: VT [B][KVH][64][T] (K-contiguous for PV)
__global__ void v_proj_kernel(const __bf16* __restrict__ Ckv,
                              const __bf16* __restrict__ WuvT,  // [128][128]
                              __bf16* __restrict__ VT) {
  int wid  = blockIdx.x * (blockDim.x >> 5) + (threadIdx.x >> 5);
  int lane = threadIdx.x & 31;
  if (wid >= (MR / 16) * KVHc) return;
  int kvh = wid % KVHc, rt = wid / KVHc;
  int mbase = rt * 16;

  v8f zero = {0.f,0.f,0.f,0.f,0.f,0.f,0.f,0.f};
  v8f a0 = zero, a1 = zero, a2 = zero, a3 = zero;
  for (int kc = 0; kc < 128; kc += 32) {
    v16bf a = load_frag(Ckv, 128, mbase, kc, lane);
    a0 = wmma_bf16(a, load_frag(WuvT, 128, kvh * 64,      kc, lane), a0);
    a1 = wmma_bf16(a, load_frag(WuvT, 128, kvh * 64 + 16, kc, lane), a1);
    a2 = wmma_bf16(a, load_frag(WuvT, 128, kvh * 64 + 32, kc, lane), a2);
    a3 = wmma_bf16(a, load_frag(WuvT, 128, kvh * 64 + 48, kc, lane), a3);
  }
  int b = mbase / Tc, t0 = mbase % Tc;
  __bf16* dst = VT + (size_t)(b * KVHc + kvh) * 64 * Tc;
  int c = lane & 15, half = (lane < 16) ? 0 : 8;
#pragma unroll
  for (int j = 0; j < 8; ++j) {
    int t = t0 + j + half;
    dst[(size_t)( 0 + c) * Tc + t] = f2bf(a0[j]);
    dst[(size_t)(16 + c) * Tc + t] = f2bf(a1[j]);
    dst[(size_t)(32 + c) * Tc + t] = f2bf(a2[j]);
    dst[(size_t)(48 + c) * Tc + t] = f2bf(a3[j]);
  }
}

// generic C[M,N] = A[M,K] @ B[K,N] with BT = B transposed [N][K]; f32 out
template <int N, int K>
__global__ void gemm_bf16_kernel(const __bf16* __restrict__ A,
                                 const __bf16* __restrict__ BT,
                                 float* __restrict__ C, int M) {
  int wid  = blockIdx.x * (blockDim.x >> 5) + (threadIdx.x >> 5);
  int lane = threadIdx.x & 31;
  constexpr int NT = N / 16;
  int mt = wid / NT, nt = wid % NT;
  if (mt * 16 >= M) return;

  v8f acc = {0.f,0.f,0.f,0.f,0.f,0.f,0.f,0.f};
  for (int kc = 0; kc < K; kc += 32)
    acc = wmma_bf16(load_frag(A, K, mt * 16, kc, lane),
                    load_frag(BT, K, nt * 16, kc, lane), acc);

  int c = lane & 15, half = (lane < 16) ? 0 : 8;
#pragma unroll
  for (int j = 0; j < 8; ++j)
    C[(size_t)(mt * 16 + j + half) * N + nt * 16 + c] = acc[j];
}

// flash-style causal attention + value-residual bypass; AO bf16 [B*T, 512]
__global__ void attn_kernel(const __bf16* __restrict__ Q,
                            const __bf16* __restrict__ Km,
                            const __bf16* __restrict__ VT,
                            const float* __restrict__ vres,
                            __bf16* __restrict__ AO) {
  __shared__ __align__(16) __bf16 Plds[4][16 * 32];
  int wib  = threadIdx.x >> 5;
  int wid  = blockIdx.x * (blockDim.x >> 5) + wib;
  int lane = threadIdx.x & 31;
  const int QT = Tc / 16;
  if (wid >= Bc * Hc * QT) return;
  int qt = wid % QT, bh = wid / QT;
  int h = bh % Hc, b = bh / Hc;
  int kvh = h >> 2;  // GQA group = H/KVH = 4

  const __bf16* Qh = Q  + ((size_t)(b * Hc   + h)   * Tc) * 64;
  const __bf16* Kh = Km + ((size_t)(b * KVHc + kvh) * Tc) * 64;
  const __bf16* Vh = VT + (size_t)(b * KVHc + kvh) * 64 * Tc;
  int qbase = qt * 16;

  v16bf qa0 = load_frag(Qh, 64, qbase, 0,  lane);
  v16bf qa1 = load_frag(Qh, 64, qbase, 32, lane);

  v8f zero = {0.f,0.f,0.f,0.f,0.f,0.f,0.f,0.f};
  v8f o0 = zero, o1 = zero, o2 = zero, o3 = zero;
  float m[8], l[8];
#pragma unroll
  for (int j = 0; j < 8; ++j) { m[j] = -1e30f; l[j] = 0.f; }

  int c = lane & 15, half = (lane < 16) ? 0 : 8;
  const float scale = 0.125f;  // 64^-0.5

  for (int kb = 0; kb < qbase + 16; kb += 32) {
    __builtin_prefetch(Kh + (size_t)(kb + 32) * 64, 0, 1);  // global_prefetch_b8
    v8f s0 = zero, s1 = zero;
    s0 = wmma_bf16(qa0, load_frag_clamp(Kh, 64, kb,      0,  Tc - 1, lane), s0);
    s0 = wmma_bf16(qa1, load_frag_clamp(Kh, 64, kb,      32, Tc - 1, lane), s0);
    s1 = wmma_bf16(qa0, load_frag_clamp(Kh, 64, kb + 16, 0,  Tc - 1, lane), s1);
    s1 = wmma_bf16(qa1, load_frag_clamp(Kh, 64, kb + 16, 32, Tc - 1, lane), s1);

    int k0 = kb + c, k1 = kb + 16 + c;
#pragma unroll
    for (int j = 0; j < 8; ++j) {
      int qr = qbase + j + half;
      float v0 = (k0 <= qr) ? s0[j] * scale : -1e30f;
      float v1 = (k1 <= qr) ? s1[j] * scale : -1e30f;
      float tm = red_max16(fmaxf(v0, v1));
      float mn = fmaxf(m[j], tm);
      float corr = __expf(m[j] - mn);
      float p0 = __expf(v0 - mn);
      float p1 = __expf(v1 - mn);
      float rs = red_add16(p0 + p1);
      l[j] = l[j] * corr + rs;
      m[j] = mn;
      o0[j] *= corr; o1[j] *= corr; o2[j] *= corr; o3[j] *= corr;
      int mrow = j + half;
      Plds[wib][mrow * 32 + c]      = f2bf(p0);   // C-layout -> row-major P
      Plds[wib][mrow * 32 + 16 + c] = f2bf(p1);
    }
    // per-wave LDS ordering guarantees store->load order within the wave
    v16bf ap = load_frag(&Plds[wib][0], 32, 0, 0, lane);
    o0 = wmma_bf16(ap, load_frag(Vh, Tc,  0, kb, lane), o0);
    o1 = wmma_bf16(ap, load_frag(Vh, Tc, 16, kb, lane), o1);
    o2 = wmma_bf16(ap, load_frag(Vh, Tc, 32, kb, lane), o2);
    o3 = wmma_bf16(ap, load_frag(Vh, Tc, 48, kb, lane), o3);
  }

#pragma unroll
  for (int j = 0; j < 8; ++j) {
    float inv = 1.0f / l[j];
    int t = qbase + j + half;
    size_t rb = ((size_t)b * Tc + t) * 512 + h * 64;
    AO[rb +  0 + c] = f2bf(o0[j] * inv + vres[rb +  0 + c]);
    AO[rb + 16 + c] = f2bf(o1[j] * inv + vres[rb + 16 + c]);
    AO[rb + 32 + c] = f2bf(o2[j] * inv + vres[rb + 32 + c]);
    AO[rb + 48 + c] = f2bf(o3[j] * inv + vres[rb + 48 + c]);
  }
}

// ---------------- launcher ----------------
extern "C" void kernel_launch(void* const* d_in, const int* in_sizes, int n_in,
                              void* d_out, int out_size, void* d_ws, size_t ws_size,
                              hipStream_t stream) {
  (void)n_in; (void)out_size; (void)ws_size; (void)in_sizes;
  const float* x_f    = (const float*)d_in[0];
  const float* wdq_f  = (const float*)d_in[1];
  const float* wuq_f  = (const float*)d_in[2];
  const float* wqr_f  = (const float*)d_in[3];
  const float* wdkv_f = (const float*)d_in[4];
  const float* wuk_f  = (const float*)d_in[5];
  const float* wuv_f  = (const float*)d_in[6];
  const float* wkr_f  = (const float*)d_in[7];
  const float* wvr_f  = (const float*)d_in[8];
  const float* wo_f   = (const float*)d_in[9];
  const float* q_ln   = (const float*)d_in[10];
  const float* kv_ln  = (const float*)d_in[11];
  const float* q_hw   = (const float*)d_in[12];
  const float* k_hw   = (const float*)d_in[13];

  // carve workspace (~54 MB total, 256B-aligned regions with slack)
  char* p = (char*)d_ws;
  auto carve = [&](size_t bytes) {
    void* r = (void*)p;
    p += (bytes + 511) & ~(size_t)255;
    return r;
  };
  __bf16* x_bf   = (__bf16*)carve((size_t)MR * 512 * 2);
  __bf16* wdqT   = (__bf16*)carve(512 * 256 * 2);
  __bf16* wuqT   = (__bf16*)carve(256 * 256 * 2);
  __bf16* wqrT   = (__bf16*)carve(256 * 256 * 2);
  __bf16* wdkvT  = (__bf16*)carve(512 * 128 * 2);
  __bf16* wukT   = (__bf16*)carve(128 * 64 * 2);
  __bf16* wuvT   = (__bf16*)carve(128 * 128 * 2);
  __bf16* wkrT   = (__bf16*)carve(512 * 64 * 2);
  __bf16* wvrT   = (__bf16*)carve(512 * 512 * 2);
  __bf16* woT    = (__bf16*)carve(512 * 512 * 2);
  __bf16* c_q    = (__bf16*)carve((size_t)MR * 256 * 2);
  __bf16* c_kv   = (__bf16*)carve((size_t)MR * 128 * 2);
  __bf16* q_buf  = (__bf16*)carve((size_t)Bc * Hc * Tc * 64 * 2);
  __bf16* k_buf  = (__bf16*)carve((size_t)Bc * KVHc * Tc * 64 * 2);
  __bf16* vt_buf = (__bf16*)carve((size_t)Bc * KVHc * 64 * Tc * 2);
  float*  vres   = (float*) carve((size_t)MR * 512 * 4);
  __bf16* ao_bf  = (__bf16*)carve((size_t)MR * 512 * 2);

  // 1) stage: x cast, weights cast+transpose
  cast_f32_to_bf16<<<2048, 256, 0, stream>>>(x_f, x_bf, MR * 512);
  transpose_cast_kernel<<<512, 256, 0, stream>>>(wdq_f,  wdqT,  512, 256);
  transpose_cast_kernel<<<256, 256, 0, stream>>>(wuq_f,  wuqT,  256, 256);
  transpose_cast_kernel<<<256, 256, 0, stream>>>(wqr_f,  wqrT,  256, 256);
  transpose_cast_kernel<<<256, 256, 0, stream>>>(wdkv_f, wdkvT, 512, 128);
  transpose_cast_kernel<<<32,  256, 0, stream>>>(wuk_f,  wukT,  128, 64);
  transpose_cast_kernel<<<64,  256, 0, stream>>>(wuv_f,  wuvT,  128, 128);
  transpose_cast_kernel<<<128, 256, 0, stream>>>(wkr_f,  wkrT,  512, 64);
  transpose_cast_kernel<<<1024,256, 0, stream>>>(wvr_f,  wvrT,  512, 512);
  transpose_cast_kernel<<<1024,256, 0, stream>>>(wo_f,   woT,   512, 512);

  // 2) down-proj + RMSNorm
  rms_down_kernel<256><<<128, 128, 0, stream>>>(x_bf, wdqT,  q_ln,  c_q);
  rms_down_kernel<128><<<128, 128, 0, stream>>>(x_bf, wdkvT, kv_ln, c_kv);

  // 3) up-proj + RoPE + head RMSNorm
  q_proj_kernel<<<1024, 128, 0, stream>>>(c_q, wuqT, wqrT, q_hw, q_buf);
  k_proj_kernel<<<256, 128, 0, stream>>>(c_kv, x_bf, wukT, wkrT, k_hw, k_buf);
  v_proj_kernel<<<256, 128, 0, stream>>>(c_kv, wuvT, vt_buf);

  // 4) value residual: v_res = x @ W_VR
  gemm_bf16_kernel<512, 512><<<4096, 128, 0, stream>>>(x_bf, wvrT, vres, MR);

  // 5) causal attention + residual
  attn_kernel<<<1024, 128, 0, stream>>>(q_buf, k_buf, vt_buf, vres, ao_bf);

  // 6) out = (attn + v_res) @ W_O -> fp32 d_out
  gemm_bf16_kernel<512, 512><<<4096, 128, 0, stream>>>(ao_bf, woT, (float*)d_out, MR);
}
